// Model_GIN_2834678415940
// MI455X (gfx1250) — matrix operand order repaired and checked by reading, not compiled, not run on hardware
//
#include <hip/hip_runtime.h>
#include <hip/hip_bf16.h>
#include <cstddef>

// ---------------------------------------------------------------------------
// Sizes from the reference
// ---------------------------------------------------------------------------
#define NTOT   16384      // B*N
#define TIME_L 1024
#define FFT_NP 1024
#define FFT_H  513        // FFT_N/2+1
#define FEATW  1026       // 2*FFT_H (logical)
#define FEATS  1056       // padded row stride (33*32) so the GIN GEMM is regular
#define NEDGE  262144
#define NBAT   64
#define NNODE  256
#define NHEAD  4
#define MODEN  8
#define DIMN   128
#define DFFN   256
#define OSTR   32         // attention-output row stride (wo GEMM K padded 16->32)

typedef __bf16 bf16_t;
typedef __attribute__((ext_vector_type(16))) __bf16 v16bf;
typedef __attribute__((ext_vector_type(8)))  float  v8f;

union PkBf4 { bf16_t h[4]; unsigned long long u64; };

// ---------------------------------------------------------------------------
// bf16-WMMA GEMM:  C[M,N] = act(A[M,K] @ W[K,N] + bias)
// All call sites guarantee M%128==0, N%BNT==0, K%32==0, 16B-aligned rows
// (buffers are padded on the host side), so there is NO guarded path:
// float4 global loads + register double-buffering hide HBM latency behind
// the WMMA pipe.  Tile: BM=128 x BNT(64|16), BK=32; 8 waves; wave w owns rows
// w*16..+15 with BNT/16 n-subtiles.   ACT: 0 = none, 1 = relu, 2 = tanh
// ---------------------------------------------------------------------------
template<int ACT, int BNT>
__global__ __launch_bounds__(256)
void gemm_kernel(const float* __restrict__ A, const float* __restrict__ W,
                 const float* __restrict__ bias, float* __restrict__ C,
                 int M, int N, int K)
{
    constexpr int BM = 128, BK = 32;
    constexpr int NS = BNT / 16;              // n-subtiles per wave
    constexpr int BI = (BNT * 8 + 255) / 256; // float4 B-loads per thread
    __shared__ bf16_t sA[BM][BK];             // 8 KB
    __shared__ bf16_t sB[BNT][BK];            // 4 KB (64) / 1 KB (16), transposed

    const int tile_m = blockIdx.y * BM;
    const int tile_n = blockIdx.x * BNT;
    const int tid  = threadIdx.x;
    const int wave = tid >> 5;
    const int lane = tid & 31;
    const int hi   = lane >> 4;
    const int l15  = lane & 15;

    v8f acc[NS];
#pragma unroll
    for (int t = 0; t < NS; ++t)
#pragma unroll
        for (int i = 0; i < 8; ++i) acc[t][i] = 0.f;

    // A: 1024 float4 per tile -> 4 per thread (row = idx>>3, kk = (idx&7)*4)
    // B: BNT*8 float4 per tile -> BI per thread (kk = idx/(BNT/4), n4 = idx%(BNT/4))
    float4 aF[4];
    float4 bF[BI];

    auto ldA = [&](int kc) {
#pragma unroll
        for (int i = 0; i < 4; ++i) {
            const int idx = tid + i * 256;
            const int row = idx >> 3;
            const int kk  = (idx & 7) << 2;
            aF[i] = *reinterpret_cast<const float4*>(
                &A[(size_t)(tile_m + row) * K + kc + kk]);
        }
    };
    auto stA = [&]() {
#pragma unroll
        for (int i = 0; i < 4; ++i) {
            const int idx = tid + i * 256;
            const int row = idx >> 3;
            const int kk  = (idx & 7) << 2;
            PkBf4 p;
            p.h[0] = (bf16_t)aF[i].x; p.h[1] = (bf16_t)aF[i].y;
            p.h[2] = (bf16_t)aF[i].z; p.h[3] = (bf16_t)aF[i].w;
            *reinterpret_cast<unsigned long long*>(&sA[row][kk]) = p.u64;
        }
    };
    auto ldB = [&](int kc) {
#pragma unroll
        for (int i = 0; i < BI; ++i) {
            const int idx = tid + i * 256;
            if (idx < BNT * 8) {
                const int kk = idx / (BNT / 4);
                const int n4 = idx % (BNT / 4);
                bF[i] = *reinterpret_cast<const float4*>(
                    &W[(size_t)(kc + kk) * N + tile_n + n4 * 4]);
            }
        }
    };
    auto stB = [&]() {
#pragma unroll
        for (int i = 0; i < BI; ++i) {
            const int idx = tid + i * 256;
            if (idx < BNT * 8) {
                const int kk = idx / (BNT / 4);
                const int n4 = idx % (BNT / 4);
                sB[n4 * 4 + 0][kk] = (bf16_t)bF[i].x;
                sB[n4 * 4 + 1][kk] = (bf16_t)bF[i].y;
                sB[n4 * 4 + 2][kk] = (bf16_t)bF[i].z;
                sB[n4 * 4 + 3][kk] = (bf16_t)bF[i].w;
            }
        }
    };

    ldA(0); ldB(0);
    for (int k0 = 0; k0 < K; k0 += BK) {
        stA(); stB();
        __syncthreads();
        if (k0 + BK < K) { ldA(k0 + BK); ldB(k0 + BK); }  // overlap with WMMAs

        // A fragment (16-bit A 16x32): lanes 0-15 row=l K={0..7,16..23};
        // lanes 16-31 row=l-16 K={8..15,24..31}
        v16bf afrag;
        {
            const bf16_t* ap = &sA[(wave << 4) + l15][0];
            const int kb0 = hi ? 8 : 0;
            const int kb1 = hi ? 24 : 16;
#pragma unroll
            for (int i = 0; i < 8; ++i) afrag[i] = ap[kb0 + i];
#pragma unroll
            for (int i = 0; i < 8; ++i) afrag[8 + i] = ap[kb1 + i];
        }
#pragma unroll
        for (int nt = 0; nt < NS; ++nt) {
            v16bf bfrag;
            const bf16_t* bp = &sB[nt * 16 + l15][hi ? 8 : 0];
#pragma unroll
            for (int i = 0; i < 8; ++i) bfrag[i] = bp[i];
#pragma unroll
            for (int i = 0; i < 8; ++i) bfrag[8 + i] = bp[16 + i];
            acc[nt] = __builtin_amdgcn_wmma_f32_16x16x32_bf16(
                false, afrag, false, bfrag, (short)0, acc[nt], false, false);
        }
        __syncthreads();
    }

    // ---- epilogue (C/D layout: VGPR v -> M = v + 8*hi, N = lane&15)
#pragma unroll
    for (int nt = 0; nt < NS; ++nt) {
        const int gn = tile_n + nt * 16 + l15;
        const float bv = bias ? bias[gn] : 0.f;
#pragma unroll
        for (int v = 0; v < 8; ++v) {
            const int gm = tile_m + (wave << 4) + v + (hi << 3);
            float val = acc[nt][v] + bv;
            if (ACT == 1) val = fmaxf(val, 0.f);
            else if (ACT == 2) val = tanhf(val);
            C[(size_t)gm * N + gn] = val;
        }
    }
}

// ---------------------------------------------------------------------------
// Pad/copy a weight matrix: Wdst[Kp,Np] = Wsrc[Ks,Ns] zero-extended.
// Also used for 1-row bias padding.
// ---------------------------------------------------------------------------
__global__ void pad_weight(const float* __restrict__ Wsrc, float* __restrict__ Wdst,
                           int Ks, int Ns, int Kp, int Np)
{
    const int i = blockIdx.x * 256 + threadIdx.x;
    if (i < Kp * Np) {
        const int k = i / Np, n = i % Np;
        Wdst[i] = (k < Ks && n < Ns) ? Wsrc[(size_t)k * Ns + n] : 0.f;
    }
}

__global__ void zero_buf(float* __restrict__ p, int n)
{
    const int i = blockIdx.x * 256 + threadIdx.x;
    if (i < n) p[i] = 0.f;
}

// ---------------------------------------------------------------------------
// 1024-point radix-2 FFT of each row (real input) -> feat = [|F|, angle(F)],
// row stride FEATS with zeroed pad columns (FEATW..FEATS).
// ---------------------------------------------------------------------------
__global__ __launch_bounds__(256)
void fft_feat_kernel(const float* __restrict__ x, float* __restrict__ feat)
{
    __shared__ float re[FFT_NP];
    __shared__ float im[FFT_NP];
    const size_t row = blockIdx.x;
    const int tid = threadIdx.x;

#pragma unroll
    for (int i = 0; i < 4; ++i) {
        const int idx = tid + i * 256;
        const int r = (int)(__builtin_bitreverse32((unsigned)idx) >> 22);  // 10-bit reverse
        re[r] = x[row * FFT_NP + idx];
        im[r] = 0.f;
    }
    __syncthreads();

    for (int s = 1; s <= 10; ++s) {
        const int len = 1 << s, half = len >> 1;
        for (int t = tid; t < FFT_NP / 2; t += 256) {
            const int grp = t / half, pos = t % half;
            const int i0 = grp * len + pos, i1 = i0 + half;
            const float ang = -3.14159265358979f * (float)pos / (float)half;
            float si, co;
            sincosf(ang, &si, &co);
            const float vr = re[i1], vi = im[i1];
            const float tr = vr * co - vi * si;
            const float ti = vr * si + vi * co;
            const float ur = re[i0], ui = im[i0];
            re[i0] = ur + tr; im[i0] = ui + ti;
            re[i1] = ur - tr; im[i1] = ui - ti;
        }
        __syncthreads();
    }

    for (int k = tid; k < FFT_H; k += 256) {
        const float a = re[k], b = im[k];
        feat[row * FEATS + k]          = sqrtf(a * a + b * b);
        feat[row * FEATS + FFT_H + k]  = atan2f(b, a);
    }
    if (tid < FEATS - FEATW)                       // zero the pad columns
        feat[row * FEATS + FEATW + tid] = 0.f;
}

// ---------------------------------------------------------------------------
// Edge scatter-add:  agg[dst] += feat[src]   (one block per edge, stride FEATS)
// ---------------------------------------------------------------------------
__global__ __launch_bounds__(256)
void scatter_kernel(const float* __restrict__ feat, const int* __restrict__ src,
                    const int* __restrict__ dst, float* __restrict__ agg)
{
    const int e = blockIdx.x;
    const float* fp = feat + (size_t)src[e] * FEATS;
    float*       ap = agg  + (size_t)dst[e] * FEATS;
    for (int j = threadIdx.x; j < FEATW; j += 256)
        atomicAdd(ap + j, fp[j]);
}

// ---------------------------------------------------------------------------
// Per-(batch,head) attention: Lq query rows, Lm kv rows, head dim 4.
// Two-pass online softmax; K/V staged in LDS. scale = 1/sqrt(4) = 0.5.
// Q/K/V have row stride 16; O has row stride OSTR (=32, pre-zeroed pad).
// ---------------------------------------------------------------------------
__global__ __launch_bounds__(256)
void attn_kernel(const float* __restrict__ Q, const float* __restrict__ K,
                 const float* __restrict__ V, float* __restrict__ O,
                 int Lq, int Lm)
{
    const int b = blockIdx.x >> 2;
    const int h = blockIdx.x & 3;
    __shared__ float sk[NNODE * 4];
    __shared__ float sv[NNODE * 4];
    const int tid = threadIdx.x;

    for (int idx = tid; idx < Lm * 4; idx += 256) {
        const int j = idx >> 2, d = idx & 3;
        const size_t g = ((size_t)b * Lm + j) * 16 + h * 4 + d;
        sk[idx] = K[g];
        sv[idx] = V[g];
    }
    __syncthreads();

    if (tid < Lq) {
        const size_t qb = ((size_t)b * Lq + tid) * 16 + h * 4;
        const size_t ob = ((size_t)b * Lq + tid) * OSTR + h * 4;
        const float q0 = Q[qb], q1 = Q[qb + 1], q2 = Q[qb + 2], q3 = Q[qb + 3];
        float mx = -1e30f;
        for (int j = 0; j < Lm; ++j) {
            const float* kp = sk + j * 4;
            const float s = 0.5f * (q0 * kp[0] + q1 * kp[1] + q2 * kp[2] + q3 * kp[3]);
            mx = fmaxf(mx, s);
        }
        float den = 0.f, a0 = 0.f, a1 = 0.f, a2 = 0.f, a3 = 0.f;
        for (int j = 0; j < Lm; ++j) {
            const float* kp = sk + j * 4;
            const float* vp = sv + j * 4;
            const float s = 0.5f * (q0 * kp[0] + q1 * kp[1] + q2 * kp[2] + q3 * kp[3]);
            const float e = __expf(s - mx);
            den += e; a0 += e * vp[0]; a1 += e * vp[1]; a2 += e * vp[2]; a3 += e * vp[3];
        }
        const float inv = 1.f / den;
        O[ob] = a0 * inv; O[ob + 1] = a1 * inv; O[ob + 2] = a2 * inv; O[ob + 3] = a3 * inv;
    }
}

// ---------------------------------------------------------------------------
// out[row] = LayerNorm(A[row] + Bs[row]) * gam + bet    (D = 1024, in-place OK)
// ---------------------------------------------------------------------------
__global__ __launch_bounds__(256)
void ln_kernel(const float* __restrict__ A, const float* __restrict__ Bs,
               const float* __restrict__ gam, const float* __restrict__ bet,
               float* __restrict__ out)
{
    const int D = TIME_L;
    const size_t row = blockIdx.x;
    __shared__ float red[256];
    const int tid = threadIdx.x;

    float xv[4];
    float s = 0.f;
#pragma unroll
    for (int i = 0; i < 4; ++i) {
        const int idx = tid + i * 256;
        xv[i] = A[row * D + idx] + Bs[row * D + idx];
        s += xv[i];
    }
    red[tid] = s; __syncthreads();
    for (int off = 128; off > 0; off >>= 1) {
        if (tid < off) red[tid] += red[tid + off];
        __syncthreads();
    }
    const float mean = red[0] * (1.f / 1024.f);
    __syncthreads();
    float s2 = 0.f;
#pragma unroll
    for (int i = 0; i < 4; ++i) { const float d = xv[i] - mean; s2 += d * d; }
    red[tid] = s2; __syncthreads();
    for (int off = 128; off > 0; off >>= 1) {
        if (tid < off) red[tid] += red[tid + off];
        __syncthreads();
    }
    const float rstd = rsqrtf(red[0] * (1.f / 1024.f) + 1e-5f);
#pragma unroll
    for (int i = 0; i < 4; ++i) {
        const int idx = tid + i * 256;
        out[row * D + idx] = (xv[i] - mean) * rstd * gam[idx] + bet[idx];
    }
}

// ---------------------------------------------------------------------------
// Broadcast PMA seeds (8,1024) -> (64*8,1024)
// ---------------------------------------------------------------------------
__global__ void bcast_seeds(const float* __restrict__ seeds, float* __restrict__ out)
{
    const int i = blockIdx.x * 256 + threadIdx.x;
    if (i < 512 * 1024) {
        const int c = i & 1023;
        const int r = (i >> 10) & 7;
        out[i] = seeds[r * 1024 + c];
    }
}

// ---------------------------------------------------------------------------
// phi column-max-abs normalize.  phiPad has row stride 16 (cols 0..7 valid);
// final output is written densely (16384,8).
// ---------------------------------------------------------------------------
__global__ void colmax_init(unsigned* cm) { if (threadIdx.x < 8) cm[threadIdx.x] = 0u; }

__global__ __launch_bounds__(256)
void colmax_kernel(const float* __restrict__ phiPad, unsigned* __restrict__ cm)
{
    const int r = blockIdx.x * 256 + threadIdx.x;      // 64 blocks -> exactly NTOT rows
    float lm[8];
#pragma unroll
    for (int c = 0; c < 8; ++c) lm[c] = 0.f;
    if (r < NTOT) {
#pragma unroll
        for (int c = 0; c < 8; ++c) lm[c] = fabsf(phiPad[(size_t)r * 16 + c]);
    }
#pragma unroll
    for (int c = 0; c < 8; ++c)
        atomicMax(&cm[c], __float_as_uint(lm[c]));     // nonneg float bits order as uints
}

__global__ __launch_bounds__(256)
void colmax_div(const float* __restrict__ phiPad, const unsigned* __restrict__ cm,
                float* __restrict__ out_phi)
{
    const int i = blockIdx.x * 256 + threadIdx.x;
    if (i < NTOT * 8) {
        const int r = i >> 3, c = i & 7;
        out_phi[i] = phiPad[(size_t)r * 16 + c] / __uint_as_float(cm[c]);
    }
}

// ---------------------------------------------------------------------------
// Host-side orchestration
// ---------------------------------------------------------------------------
struct MhaP { const float* p[12]; };   // wq wk wv wo l1g l1b fw1 fb1 fw2 fb2 l2g l2b

static MhaP mha_from(void* const* d_in, int base) {
    MhaP m;
    for (int i = 0; i < 12; ++i) m.p[i] = (const float*)d_in[base + i];
    return m;
}

static void gemm(hipStream_t st, int act, const float* A, const float* W,
                 const float* bias, float* C, int M, int N, int K)
{
    dim3 blk(256);
    if (N <= 16) {
        dim3 g(N / 16, M / 128);
        if (act == 0)      gemm_kernel<0,16><<<g, blk, 0, st>>>(A, W, bias, C, M, N, K);
        else if (act == 1) gemm_kernel<1,16><<<g, blk, 0, st>>>(A, W, bias, C, M, N, K);
        else               gemm_kernel<2,16><<<g, blk, 0, st>>>(A, W, bias, C, M, N, K);
    } else {
        dim3 g(N / 64, M / 128);
        if (act == 0)      gemm_kernel<0,64><<<g, blk, 0, st>>>(A, W, bias, C, M, N, K);
        else if (act == 1) gemm_kernel<1,64><<<g, blk, 0, st>>>(A, W, bias, C, M, N, K);
        else               gemm_kernel<2,64><<<g, blk, 0, st>>>(A, W, bias, C, M, N, K);
    }
}

// One SAB / cross-attention block, DGL style (post-norm residuals).
//   TA: rowsq x 1024 scratch (holds o then x1)   TB: rowsq x 1024 scratch (ff)
//   OUT may alias TB (ln is in-place safe).   woPad: 32x1024 scratch.
static void run_mha(hipStream_t st, const float* X, const float* KV,
                    int rowsq, int rowskv, int Lq, int Lm, const MhaP& m,
                    float* Qb, float* Kb, float* Vb, float* Ob, float* woPad,
                    float* TA, float* T2buf, float* TB, float* OUT)
{
    pad_weight<<<(32 * 1024 + 255) / 256, 256, 0, st>>>(m.p[3], woPad, 16, 1024, 32, 1024);
    gemm(st, 0, X,  m.p[0], nullptr, Qb, rowsq,  16, TIME_L);          // Q
    gemm(st, 0, KV, m.p[1], nullptr, Kb, rowskv, 16, TIME_L);          // K
    gemm(st, 0, KV, m.p[2], nullptr, Vb, rowskv, 16, TIME_L);          // V
    attn_kernel<<<NBAT * NHEAD, 256, 0, st>>>(Qb, Kb, Vb, Ob, Lq, Lm);
    gemm(st, 0, Ob, woPad, nullptr, TA, rowsq, TIME_L, OSTR);          // o = O @ wo
    ln_kernel<<<rowsq, 256, 0, st>>>(X, TA, m.p[4], m.p[5], TA);       // x1 = LN(X+o)
    gemm(st, 1, TA, m.p[6], m.p[7], T2buf, rowsq, DFFN, TIME_L);       // relu(x1@fw1+b)
    gemm(st, 0, T2buf, m.p[8], m.p[9], TB, rowsq, TIME_L, DFFN);       // ff
    ln_kernel<<<rowsq, 256, 0, st>>>(TA, TB, m.p[10], m.p[11], OUT);   // LN(x1+ff)
}

extern "C" void kernel_launch(void* const* d_in, const int* in_sizes, int n_in,
                              void* d_out, int out_size, void* d_ws, size_t ws_size,
                              hipStream_t stream)
{
    (void)in_sizes; (void)n_in; (void)out_size; (void)ws_size;

    const float* x   = (const float*)d_in[0];
    const int*   src = (const int*)d_in[1];
    const int*   dst = (const int*)d_in[2];
    MhaP enc0 = mha_from(d_in, 3);
    MhaP enc1 = mha_from(d_in, 15);
    const float* seeds  = (const float*)d_in[27];
    const float* rff_w1 = (const float*)d_in[28];
    const float* rff_b1 = (const float*)d_in[29];
    const float* rff_w2 = (const float*)d_in[30];
    const float* rff_b2 = (const float*)d_in[31];
    MhaP pma  = mha_from(d_in, 32);
    MhaP dec0 = mha_from(d_in, 44);
    MhaP dec1 = mha_from(d_in, 56);
    const float* gw = (const float*)d_in[68];
    const float* gb = (const float*)d_in[69];
    const float* w1 = (const float*)d_in[70];
    const float* b1 = (const float*)d_in[71];
    const float* w2 = (const float*)d_in[72];
    const float* b2 = (const float*)d_in[73];
    const float* w3 = (const float*)d_in[74];
    const float* b3 = (const float*)d_in[75];

    // workspace layout (floats); feat/agg recycled as transformer ping-pong bufs
    float* ws = (float*)d_ws;
    size_t o = 0;
    float* feat   = ws + o; o += (size_t)NTOT * FEATS;   // also bufA (>= NTOT*1024)
    float* agg    = ws + o; o += (size_t)NTOT * FEATS;   // also bufB
    float* t2     = ws + o; o += (size_t)NTOT * DFFN;
    float* h0     = ws + o; o += (size_t)NTOT * DIMN;
    float* h1     = ws + o; o += (size_t)NTOT * DIMN;
    float* qb     = ws + o; o += (size_t)NTOT * 16;
    float* kb     = ws + o; o += (size_t)NTOT * 16;
    float* vb     = ws + o; o += (size_t)NTOT * 16;
    float* ob     = ws + o; o += (size_t)NTOT * OSTR;
    float* gS     = ws + o; o += (size_t)512 * TIME_L;
    float* g1     = ws + o; o += (size_t)512 * TIME_L;
    float* g2     = ws + o; o += (size_t)512 * TIME_L;
    float* gt2    = ws + o; o += (size_t)512 * DFFN;
    float* gwPad  = ws + o; o += (size_t)FEATS * DIMN;   // (1056,128)
    float* woPad  = ws + o; o += (size_t)32 * 1024;      // (32,1024), re-padded per layer
    float* w3Pad  = ws + o; o += (size_t)DIMN * 16;      // (128,16)
    float* b3Pad  = ws + o; o += 16;
    float* phiPad = ws + o; o += (size_t)NTOT * 16;
    unsigned* cm  = (unsigned*)(ws + o); o += 8;

    float* out_q   = (float*)d_out;                            // (64,1024,8) = g flat
    float* out_phi = out_q + (size_t)NBAT * MODEN * TIME_L;    // (16384,8)

    // ---------------- GIN branch ----------------
    fft_feat_kernel<<<NTOT, 256, 0, stream>>>(x, feat);
    pad_weight<<<((FEATS * DIMN) + 255) / 256, 256, 0, stream>>>(gw, gwPad,
                                                                 FEATW, DIMN, FEATS, DIMN);
    pad_weight<<<(DIMN * 16 + 255) / 256, 256, 0, stream>>>(w3, w3Pad, DIMN, 8, DIMN, 16);
    pad_weight<<<1, 256, 0, stream>>>(b3, b3Pad, 1, 8, 1, 16);
    hipMemcpyAsync(agg, feat, (size_t)NTOT * FEATS * sizeof(float),
                   hipMemcpyDeviceToDevice, stream);           // agg := feat (eps=0 self term)
    scatter_kernel<<<NEDGE, 256, 0, stream>>>(feat, src, dst, agg);
    gemm(stream, 0, agg, gwPad, gb, h0, NTOT, DIMN, FEATS);    // hn (K padded, pad cols = 0)
    gemm(stream, 2, h0, w1, b1, h1, NTOT, DIMN, DIMN);         // tanh layer 1
    gemm(stream, 2, h1, w2, b2, h0, NTOT, DIMN, DIMN);         // hidden 1
    gemm(stream, 2, h0, w2, b2, h1, NTOT, DIMN, DIMN);         // hidden 2
    gemm(stream, 0, h1, w3Pad, b3Pad, phiPad, NTOT, 16, DIMN); // phi (N padded 8->16)
    colmax_init<<<1, 32, 0, stream>>>(cm);
    colmax_kernel<<<NTOT / 256, 256, 0, stream>>>(phiPad, cm);
    colmax_div<<<(NTOT * 8 + 255) / 256, 256, 0, stream>>>(phiPad, cm, out_phi);

    // ---------------- set-transformer ----------------
    zero_buf<<<(NTOT * OSTR + 255) / 256, 256, 0, stream>>>(ob, NTOT * OSTR);
    // encoder SABs (feat/agg now free -> big ping-pong buffers)
    run_mha(stream, x,   x,   NTOT, NTOT, NNODE, NNODE, enc0,
            qb, kb, vb, ob, woPad, feat, t2, agg, agg);
    run_mha(stream, agg, agg, NTOT, NTOT, NNODE, NNODE, enc1,
            qb, kb, vb, ob, woPad, feat, t2, agg, agg);
    // PMA: mem = rFF(h)
    gemm(stream, 1, agg, rff_w1, rff_b1, t2,   NTOT, DFFN,   TIME_L);
    gemm(stream, 0, t2,  rff_w2, rff_b2, feat, NTOT, TIME_L, DFFN);
    bcast_seeds<<<(512 * 1024 + 255) / 256, 256, 0, stream>>>(seeds, gS);
    run_mha(stream, gS, feat, 512, NTOT, MODEN, NNODE, pma,
            qb, kb, vb, ob, woPad, g1, gt2, g2, g2);
    // decoder SABs on g (64,8,1024)
    run_mha(stream, g2, g2, 512, 512, MODEN, MODEN, dec0,
            qb, kb, vb, ob, woPad, g1, gt2, gS, gS);
    run_mha(stream, gS, gS, 512, 512, MODEN, MODEN, dec1,
            qb, kb, vb, ob, woPad, g1, gt2, g2, g2);

    // q output = g flattened (reshape is a no-op on the flat buffer)
    hipMemcpyAsync(out_q, g2, (size_t)NBAT * MODEN * TIME_L * sizeof(float),
                   hipMemcpyDeviceToDevice, stream);
}
